// GIN_28991029248650
// MI455X (gfx1250) — compile-verified, matrix operand order checked by reading, not run on hardware
//
#include <hip/hip_runtime.h>
#include <hip/hip_bf16.h>

// ---------------------------------------------------------------------------
// GIN + Set2Set forward for MI455X (gfx1250), wave32, WMMA f16->f32.
// Weights are pre-swizzled once into f16 WMMA-fragment layout so the GEMM
// inner loop is pure {b128 loads + v_wmma}; only the activation operand
// (h (+agg), produced in f32) is converted in-loop.
// ---------------------------------------------------------------------------

typedef __attribute__((ext_vector_type(16))) _Float16 v16h;
typedef __attribute__((ext_vector_type(8)))  float    v8f;

#define N_NODES  100000
#define N_EDGES  1600000
#define N_GRAPHS 512
#define NF   64
#define H1   128
#define H2   64
#define STEPS 4

__device__ __forceinline__ float gelu_f(float x) {
    return 0.5f * x * (1.0f + erff(x * 0.70710678118654752f));
}
__device__ __forceinline__ float sigmoid_f(float x) {
    return 1.0f / (1.0f + expf(-x));
}

// ---- WMMA 16-bit fragment index math (CDNA5 ISA 7.12.2) -------------------
// A (16x32): lane m = lane&15, half-wave g = lane>>4; VGPR i holds K pair at
//            k0 = (i>>2)*16 + g*8 + (i&3)*2.
// B (32x16): mirrored column-major layout, lane holds column n = lane&15.

template<bool HASAGG>
__device__ __forceinline__ v16h load_afrag(const float* X, const float* AGG,
                                           int node0, int ld, int kbase, int lane) {
    int m = lane & 15, g = lane >> 4;
    const float* xr = X + (size_t)(node0 + m) * ld + kbase;
    const float* ar = HASAGG ? (AGG + (size_t)(node0 + m) * ld + kbase) : nullptr;
    v16h a;
#pragma unroll
    for (int i = 0; i < 8; ++i) {
        int k0 = ((i >> 2) << 4) + (g << 3) + ((i & 3) << 1);
        float e0 = xr[k0], e1 = xr[k0 + 1];
        if (HASAGG) { e0 += ar[k0]; e1 += ar[k0 + 1]; }
        a[2 * i]     = (_Float16)e0;
        a[2 * i + 1] = (_Float16)e1;
    }
    return a;
}

// One-shot: convert W [NOUT, K] (f32, row-major, torch layout) into f16
// fragments: Wf[((nt*NK + kk)*32 + lane)*16 + e], e = packed VGPR element.
__global__ void k_prep_w(const float* __restrict__ W, _Float16* __restrict__ Wf,
                         int K, int NOUT) {
    int t = blockIdx.x * blockDim.x + threadIdx.x;
    if (t >= NOUT * K) return;
    int NK   = K >> 5;
    int frag = t >> 9;          // 512 halves per fragment
    int r    = t & 511;
    int lane = r >> 4, e = r & 15;
    int i = e >> 1, lo = e & 1;
    int n = lane & 15, g = lane >> 4;
    int k0 = ((i >> 2) << 4) + (g << 3) + ((i & 3) << 1) + lo;
    int nt = frag / NK, kk = frag - nt * NK;
    Wf[t] = (_Float16)W[(size_t)(nt * 16 + n) * K + kk * 32 + k0];
}

// OUT[m, n] = gelu( (X[m,:] (+AGG[m,:])) @ W.T + bias ), one wave per 16 rows.
// Wf is the pre-swizzled f16 fragment buffer.
template<int K, int NOUT, bool HASAGG>
__global__ void k_dense_wmma(const float* __restrict__ X,
                             const float* __restrict__ AGG,
                             const _Float16* __restrict__ Wf,
                             const float* __restrict__ bias,
                             float* __restrict__ OUT, int nrows) {
    int wave  = (blockIdx.x * blockDim.x + threadIdx.x) >> 5;
    int lane  = threadIdx.x & 31;
    int node0 = wave * 16;
    if (node0 >= nrows) return;

    constexpr int NK = K / 32;
    v16h afr[NK];
#pragma unroll
    for (int kk = 0; kk < NK; ++kk)
        afr[kk] = load_afrag<HASAGG>(X, AGG, node0, K, kk * 32, lane);

    const v16h* Bf = (const v16h*)Wf;   // 32B-aligned fragment slots
    int n = lane & 15, g = lane >> 4;
#pragma unroll
    for (int nt = 0; nt < NOUT / 16; ++nt) {
        v8f acc = {};
#pragma unroll
        for (int kk = 0; kk < NK; ++kk) {
            v16h b = Bf[(nt * NK + kk) * 32 + lane];
            acc = __builtin_amdgcn_wmma_f32_16x16x32_f16(
                false, afr[kk], false, b, (short)0, acc, false, false);
        }
        float bv = bias[nt * 16 + n];
#pragma unroll
        for (int j = 0; j < 8; ++j) {
            float v = gelu_f(acc[j] + bv);
            OUT[(size_t)(node0 + j + 8 * g) * NOUT + nt * 16 + n] = v;
        }
    }
}

// ---- edge aggregation: agg[dst] += h[src] (scatter, float atomics) --------
__global__ void k_edge_agg(const float* __restrict__ H,
                           const int* __restrict__ src,
                           const int* __restrict__ dst,
                           float* __restrict__ AGG,
                           long long total, int shift, int mask) {
    long long stride = (long long)gridDim.x * blockDim.x;
    for (long long t = (long long)blockIdx.x * blockDim.x + threadIdx.x;
         t < total; t += stride) {
        int e = (int)(t >> shift), f = (int)(t & mask);
        atomicAdd(&AGG[(size_t)dst[e] * (mask + 1) + f],
                  H[(size_t)src[e] * (mask + 1) + f]);
    }
}

__global__ void k_fill(float* __restrict__ p, long long n, float v) {
    long long stride = (long long)gridDim.x * blockDim.x;
    for (long long t = (long long)blockIdx.x * blockDim.x + threadIdx.x;
         t < n; t += stride) p[t] = v;
}

// ---- in-place row L2 normalize, one wave per 64-wide row ------------------
__global__ void k_norm(float* __restrict__ H, int nrows) {
    int warp = (blockIdx.x * blockDim.x + threadIdx.x) >> 5;
    int lane = threadIdx.x & 31;
    if (warp >= nrows) return;
    float* row = H + (size_t)warp * H2;
    float v0 = row[lane], v1 = row[lane + 32];
    float ss = v0 * v0 + v1 * v1;
#pragma unroll
    for (int off = 16; off; off >>= 1) ss += __shfl_xor(ss, off, 32);
    float inv = 1.0f / fmaxf(sqrtf(ss), 1e-12f);
    row[lane] = v0 * inv;
    row[lane + 32] = v1 * inv;
}

// ---- LSTM cell (torch gate order i,f,g,o), out-of-place state -------------
__global__ void k_lstm(const float* __restrict__ X,
                       const float* __restrict__ Hin,
                       const float* __restrict__ Cin,
                       float* __restrict__ Hout, float* __restrict__ Cout,
                       const float* __restrict__ Wih,
                       const float* __restrict__ Whh,
                       const float* __restrict__ bih,
                       const float* __restrict__ bhh, int in_dim) {
    int t = blockIdx.x * blockDim.x + threadIdx.x;
    if (t >= N_GRAPHS * H2) return;
    int b = t >> 6, u = t & 63;
    const float* xb = X + (size_t)b * in_dim;
    const float* hb = Hin + (size_t)b * H2;
    float g4[4];
#pragma unroll
    for (int gi = 0; gi < 4; ++gi) {
        int row = gi * H2 + u;
        float acc = bih[row] + bhh[row];
        const float* wi = Wih + (size_t)row * in_dim;
        for (int k = 0; k < in_dim; ++k) acc += wi[k] * xb[k];
        const float* wh = Whh + (size_t)row * H2;
        for (int k = 0; k < H2; ++k) acc += wh[k] * hb[k];
        g4[gi] = acc;
    }
    float i = sigmoid_f(g4[0]), f = sigmoid_f(g4[1]);
    float gg = tanhf(g4[2]),    o = sigmoid_f(g4[3]);
    float c = f * Cin[t] + i * gg;
    Cout[t] = c;
    Hout[t] = o * tanhf(c);
}

__global__ void k_init_attn(float* __restrict__ M, float* __restrict__ S,
                            float* __restrict__ R) {
    int t = blockIdx.x * blockDim.x + threadIdx.x;
    if (t < N_GRAPHS) { M[t] = -__builtin_inff(); S[t] = 0.0f; }
    if (t < N_GRAPHS * H2) R[t] = 0.0f;
}

__global__ void k_attn_e(const float* __restrict__ H,
                         const float* __restrict__ Q,
                         const int* __restrict__ batch,
                         float* __restrict__ E) {
    int n = blockIdx.x * blockDim.x + threadIdx.x;
    if (n >= N_NODES) return;
    const float* hr = H + (size_t)n * H2;
    const float* qr = Q + (size_t)batch[n] * H2;
    float acc = 0.0f;
#pragma unroll
    for (int k = 0; k < H2; ++k) acc += hr[k] * qr[k];
    E[n] = acc;
}

__global__ void k_seg_max(const float* __restrict__ E,
                          const int* __restrict__ batch,
                          float* __restrict__ M) {
    int n = blockIdx.x * blockDim.x + threadIdx.x;
    if (n >= N_NODES) return;
    atomicMax(&M[batch[n]], E[n]);
}

__global__ void k_attn_ex(float* __restrict__ E, const int* __restrict__ batch,
                          const float* __restrict__ M, float* __restrict__ S) {
    int n = blockIdx.x * blockDim.x + threadIdx.x;
    if (n >= N_NODES) return;
    float ex = expf(E[n] - M[batch[n]]);
    E[n] = ex;
    atomicAdd(&S[batch[n]], ex);
}

__global__ void k_attn_r(const float* __restrict__ H,
                         const float* __restrict__ E,
                         const int* __restrict__ batch,
                         const float* __restrict__ S,
                         float* __restrict__ R) {
    long long t = (long long)blockIdx.x * blockDim.x + threadIdx.x;
    if (t >= (long long)N_NODES * H2) return;
    int n = (int)(t >> 6), f = (int)(t & 63);
    int b = batch[n];
    float a = E[n] / (S[b] + 1e-16f);
    atomicAdd(&R[(size_t)b * H2 + f], a * H[(size_t)n * H2 + f]);
}

__global__ void k_qstar(const float* __restrict__ Q, const float* __restrict__ R,
                        float* __restrict__ QS) {
    int t = blockIdx.x * blockDim.x + threadIdx.x;
    if (t >= N_GRAPHS * 2 * H2) return;
    int b = t >> 7, j = t & 127;
    QS[t] = (j < H2) ? Q[(size_t)b * H2 + j] : R[(size_t)b * H2 + (j - H2)];
}

__global__ void k_head(const float* __restrict__ QS,
                       const float* __restrict__ fc1W, const float* __restrict__ fc1b,
                       const float* __restrict__ fc2W, const float* __restrict__ fc2b,
                       float* __restrict__ Z) {
    int b = blockIdx.x * blockDim.x + threadIdx.x;
    if (b >= N_GRAPHS) return;
    const float* q = QS + (size_t)b * (2 * H2);
    float acc2 = fc2b[0];
#pragma unroll
    for (int u = 0; u < 32; ++u) {
        float a = fc1b[u];
        const float* w = fc1W + (size_t)u * (2 * H2);
        for (int k = 0; k < 2 * H2; ++k) a += w[k] * q[k];
        acc2 += fc2W[u] * gelu_f(a);
    }
    Z[b] = acc2;
}

// ---------------------------------------------------------------------------
extern "C" void kernel_launch(void* const* d_in, const int* in_sizes, int n_in,
                              void* d_out, int out_size, void* d_ws, size_t ws_size,
                              hipStream_t stream) {
    (void)in_sizes; (void)n_in; (void)out_size; (void)ws_size;

    const float* x      = (const float*)d_in[0];
    const int*   eidx   = (const int*)d_in[1];
    const int*   batch  = (const int*)d_in[2];
    const float* nfcW   = (const float*)d_in[3];
    const float* nfcB   = (const float*)d_in[4];
    const float* gc1W   = (const float*)d_in[5];
    const float* gc1B   = (const float*)d_in[6];
    const float* gc2W   = (const float*)d_in[7];
    const float* gc2B   = (const float*)d_in[8];
    const float* l0Wih  = (const float*)d_in[9];
    const float* l0Whh  = (const float*)d_in[10];
    const float* l0bih  = (const float*)d_in[11];
    const float* l0bhh  = (const float*)d_in[12];
    const float* l1Wih  = (const float*)d_in[13];
    const float* l1Whh  = (const float*)d_in[14];
    const float* l1bih  = (const float*)d_in[15];
    const float* l1bhh  = (const float*)d_in[16];
    const float* fc1W   = (const float*)d_in[17];
    const float* fc1B   = (const float*)d_in[18];
    const float* fc2W   = (const float*)d_in[19];
    const float* fc2B   = (const float*)d_in[20];

    const int* srcI = eidx;
    const int* dstI = eidx + N_EDGES;

    float* zOut = (float*)d_out;            // [512]
    float* hOut = zOut + N_GRAPHS;          // [100000 x 64]

    // workspace carve (floats, then f16 weight slabs at a 32B-aligned offset)
    float* ws    = (float*)d_ws;
    float* buf0  = ws;                                // 100000*128
    float* buf1  = buf0 + (size_t)N_NODES * H1;       // 100000*128
    float* eBuf  = buf1 + (size_t)N_NODES * H1;       // 100000
    float* mBuf  = eBuf + N_NODES;                    // 512
    float* sBuf  = mBuf + N_GRAPHS;                   // 512
    float* rBuf  = sBuf + N_GRAPHS;                   // 512*64
    float* qStar = rBuf + (size_t)N_GRAPHS * H2;      // 512*128
    float* st    = qStar + (size_t)N_GRAPHS * 2 * H2; // 8 * 512*64 ping-pong
    const size_t SZ = (size_t)N_GRAPHS * H2;
    float* h0p[2] = { st,            st + SZ };
    float* c0p[2] = { st + 2 * SZ,   st + 3 * SZ };
    float* h1p[2] = { st + 4 * SZ,   st + 5 * SZ };
    float* c1p[2] = { st + 6 * SZ,   st + 7 * SZ };

    _Float16* wfNfc = (_Float16*)(st + 8 * SZ);       // 128*64  halves
    _Float16* wfGc1 = wfNfc + (size_t)H1 * NF;        // 128*128 halves
    _Float16* wfGc2 = wfGc1 + (size_t)H1 * H1;        // 64*128  halves

    // 0) one-shot weight swizzle to f16 fragment layout (tiny)
    k_prep_w<<<(H1 * NF + 255) / 256, 256, 0, stream>>>(nfcW, wfNfc, NF, H1);
    k_prep_w<<<(H1 * H1 + 255) / 256, 256, 0, stream>>>(gc1W, wfGc1, H1, H1);
    k_prep_w<<<(H2 * H1 + 255) / 256, 256, 0, stream>>>(gc2W, wfGc2, H1, H2);

    const int denseBlocks = (N_NODES / 16 + 7) / 8;   // 8 waves/block, 16 rows/wave

    // 1) h1 = gelu(x @ nfc_W.T + b)          [100000 x 128]
    k_dense_wmma<NF, H1, false><<<denseBlocks, 256, 0, stream>>>(
        x, nullptr, wfNfc, nfcB, buf0, N_NODES);

    // 2) agg1 = scatter_sum(h1[src] -> dst)
    k_fill<<<2048, 256, 0, stream>>>(buf1, (long long)N_NODES * H1, 0.0f);
    k_edge_agg<<<8192, 256, 0, stream>>>(buf0, srcI, dstI, buf1,
                                         (long long)N_EDGES * H1, 7, 127);

    // 3) h2 = gelu((h1 + agg1) @ gc1_W.T + b)  (in place over buf0; each wave
    //    reads then writes only its own 16 rows)
    k_dense_wmma<H1, H1, true><<<denseBlocks, 256, 0, stream>>>(
        buf0, buf1, wfGc1, gc1B, buf0, N_NODES);

    // 4) agg2 = scatter_sum(h2[src] -> dst)
    k_fill<<<2048, 256, 0, stream>>>(buf1, (long long)N_NODES * H1, 0.0f);
    k_edge_agg<<<8192, 256, 0, stream>>>(buf0, srcI, dstI, buf1,
                                         (long long)N_EDGES * H1, 7, 127);

    // 5) h3 = gelu((h2 + agg2) @ gc2_W.T + b)  -> directly into output slab
    k_dense_wmma<H1, H2, true><<<denseBlocks, 256, 0, stream>>>(
        buf0, buf1, wfGc2, gc2B, hOut, N_NODES);

    // 6) row L2 normalize in place
    k_norm<<<(N_NODES + 7) / 8, 256, 0, stream>>>(hOut, N_NODES);

    // 7) Set2Set: zero LSTM state + q_star (ws is poisoned / stale each call)
    k_fill<<<512, 256, 0, stream>>>(qStar,
        (long long)(N_GRAPHS * 2 * H2) + (long long)(8 * SZ), 0.0f);

    int cur = 0;
    for (int s = 0; s < STEPS; ++s) {
        int nxt = 1 - cur;
        k_lstm<<<(N_GRAPHS * H2 + 255) / 256, 256, 0, stream>>>(
            qStar, h0p[cur], c0p[cur], h0p[nxt], c0p[nxt],
            l0Wih, l0Whh, l0bih, l0bhh, 2 * H2);
        k_lstm<<<(N_GRAPHS * H2 + 255) / 256, 256, 0, stream>>>(
            h0p[nxt], h1p[cur], c1p[cur], h1p[nxt], c1p[nxt],
            l1Wih, l1Whh, l1bih, l1bhh, H2);
        const float* q = h1p[nxt];

        k_init_attn<<<(N_GRAPHS * H2 + 255) / 256, 256, 0, stream>>>(mBuf, sBuf, rBuf);
        k_attn_e<<<(N_NODES + 255) / 256, 256, 0, stream>>>(hOut, q, batch, eBuf);
        k_seg_max<<<(N_NODES + 255) / 256, 256, 0, stream>>>(eBuf, batch, mBuf);
        k_attn_ex<<<(N_NODES + 255) / 256, 256, 0, stream>>>(eBuf, batch, mBuf, sBuf);
        k_attn_r<<<(N_NODES * H2 + 255) / 256, 256, 0, stream>>>(hOut, eBuf, batch, sBuf, rBuf);
        k_qstar<<<(N_GRAPHS * 2 * H2 + 255) / 256, 256, 0, stream>>>(q, rBuf, qStar);
        cur = nxt;
    }

    // 8) z = fc2(gelu(fc1(q_star)))
    k_head<<<(N_GRAPHS + 255) / 256, 256, 0, stream>>>(
        qStar, fc1W, fc1B, fc2W, fc2B, zOut);
}
